// MoveEncoder_68101001445990
// MI455X (gfx1250) — matrix operand order repaired and checked by reading, not compiled, not run on hardware
//
#include <hip/hip_runtime.h>

// MoveEncoder for MI455X (gfx1250, wave32, WMMA).
// Strategy: f16 WMMA (v_wmma_f32_16x16x32_f16) for all GEMMs, f32 accumulate.
// One wave per move; emb/Q/K staged in LDS; weights pre-converted to f16 in d_ws.
// Key algebraic cut: einsum('nlm,nmd->nd') only needs attention column sums,
// so word = colsum(att) @ V, contracted in-register against WMMA V tiles.

typedef __attribute__((ext_vector_type(16))) _Float16 v16h;
typedef __attribute__((ext_vector_type(8)))  _Float16 v8h;
typedef __attribute__((ext_vector_type(8)))  float    v8f;

#define NMOVES 32768
#define LWORD  15
#define DDIM   256
#define STRIDE 264                 // halves per LDS row (8-half pad -> no bank conflicts)
#define BUFH   (16 * STRIDE)       // halves per 16-row staging buffer
#define WPB    8                   // waves per block

// ---- prep: convert Wq/Wk/Wv f32 -> f16 into workspace (row-major kept) ----
__global__ void prep_weights_f16(const float* __restrict__ Wq,
                                 const float* __restrict__ Wk,
                                 const float* __restrict__ Wv,
                                 _Float16* __restrict__ w16) {
  int i = blockIdx.x * blockDim.x + threadIdx.x;
  if (i < DDIM * DDIM) {
    w16[i]               = (_Float16)Wq[i];
    w16[i + DDIM*DDIM]   = (_Float16)Wk[i];
    w16[i + 2*DDIM*DDIM] = (_Float16)Wv[i];
  }
}

__device__ __forceinline__ v16h cat8(v8h lo, v8h hi) {
  return __builtin_shufflevector(lo, hi, 0,1,2,3,4,5,6,7,8,9,10,11,12,13,14,15);
}

// A-tile (16x32 f16): lanes0-15 row=lane, halves0-15 -> K 0..15;
// lanes16-31 hold K 16..31 (matches B-tile order).
__device__ __forceinline__ v16h load_a_tile(const _Float16* buf, int lane, int kt) {
  const _Float16* p = buf + (lane & 15) * STRIDE + kt * 32 + ((lane >> 4) << 4);
  v8h lo = *(const v8h*)p;
  v8h hi = *(const v8h*)(p + 8);
  return cat8(lo, hi);
}

// B-tile (32x16 f16), col = lane&15; lanes0-15 hold K 0..15, lanes16-31 K 16..31.
// Source buffer holds B[k][n] = SRC[n][k]  (row-major rows of SRC are contiguous).
__device__ __forceinline__ v16h load_b_tile_lds(const _Float16* buf, int lane, int kt) {
  const _Float16* p = buf + (lane & 15) * STRIDE + kt * 32 + ((lane >> 4) << 4);
  v8h lo = *(const v8h*)p;
  v8h hi = *(const v8h*)(p + 8);
  return cat8(lo, hi);
}

__device__ __forceinline__ v16h load_b_tile_w(const _Float16* W, int lane, int nt, int kt) {
  const _Float16* p = W + (nt * 16 + (lane & 15)) * DDIM + kt * 32 + ((lane >> 4) << 4);
  v8h lo = *(const v8h*)p;
  v8h hi = *(const v8h*)(p + 8);
  return cat8(lo, hi);
}

// out[16x256] = embB[16x256] @ W^T + bias, staged to LDS as f16 (C layout scatter).
__device__ __forceinline__ void project_store(const _Float16* embB, const _Float16* W,
                                              const float* __restrict__ bias,
                                              _Float16* outB, int lane) {
  const int colb = lane & 15;
  const int rowb = (lane >> 4) << 3;
  for (int nt = 0; nt < 16; ++nt) {
    float bval = bias[nt * 16 + colb];
    v8f acc = {bval, bval, bval, bval, bval, bval, bval, bval};
#pragma unroll
    for (int kt = 0; kt < 8; ++kt) {
      v16h a = load_a_tile(embB, lane, kt);
      v16h b = load_b_tile_w(W, lane, nt, kt);
      acc = __builtin_amdgcn_wmma_f32_16x16x32_f16(false, a, false, b, (short)0, acc,
                                                   false, false);
    }
#pragma unroll
    for (int v = 0; v < 8; ++v)
      outB[(rowb + v) * STRIDE + nt * 16 + colb] = (_Float16)acc[v];
  }
}

__global__ __launch_bounds__(32 * WPB) void move_encoder_kernel(
    const int* __restrict__ words, const int* __restrict__ rows,
    const int* __restrict__ cols, const int* __restrict__ dirs,
    const int* __restrict__ scores,
    const float* __restrict__ letter_emb, const float* __restrict__ positional,
    const float* __restrict__ bq, const float* __restrict__ bk,
    const float* __restrict__ bv,
    const float* __restrict__ row_emb, const float* __restrict__ col_emb,
    const float* __restrict__ dir_emb, const float* __restrict__ score_emb,
    const _Float16* __restrict__ w16, float* __restrict__ out) {
  extern __shared__ _Float16 lds[];
  const int lane = threadIdx.x & 31;
  const int wave = threadIdx.x >> 5;
  const int move = blockIdx.x * WPB + wave;
  if (move >= NMOVES) return;  // no block-wide barriers used; safe

  _Float16* embB = lds + wave * 3 * BUFH;
  _Float16* qB   = embB + BUFH;
  _Float16* kB   = qB + BUFH;

  // ---- stage emb = letter_emb[word] + positional (f16, row 15 zero) ----
  for (int idx = lane; idx < 16 * DDIM; idx += 32) {
    int r = idx >> 8;
    int d = idx & 255;
    float v = 0.0f;
    if (r < LWORD) {
      int letter = words[move * LWORD + r];
      v = letter_emb[letter * DDIM + d] + positional[r * DDIM + d];
    }
    embB[r * STRIDE + d] = (_Float16)v;
  }
  asm volatile("s_wait_dscnt 0x0" ::: "memory");

  // ---- Q, K projections (WMMA), staged to LDS as f16 ----
  project_store(embB, w16,               bq, qB, lane);
  project_store(embB, w16 + DDIM * DDIM, bk, kB, lane);
  asm volatile("s_wait_dscnt 0x0" ::: "memory");

  // ---- S = Q @ K^T (B tiles = contiguous rows of K buffer) ----
  v8f s = {0.f, 0.f, 0.f, 0.f, 0.f, 0.f, 0.f, 0.f};
#pragma unroll
  for (int kt = 0; kt < 8; ++kt) {
    v16h a = load_a_tile(qB, lane, kt);
    v16h b = load_b_tile_lds(kB, lane, kt);
    s = __builtin_amdgcn_wmma_f32_16x16x32_f16(false, a, false, b, (short)0, s,
                                               false, false);
  }
  // mask pad key column m=15
  if ((lane & 15) == 15) {
#pragma unroll
    for (int v = 0; v < 8; ++v) s[v] = -1e30f;
  }

  // ---- row softmax + column sums. C layout: row l = v + 8*(lane>=16), col m = lane&15
  const int hi = lane >> 4;
  float colsum = 0.0f;
#pragma unroll
  for (int v = 0; v < 8; ++v) {
    float x = s[v];
    float mx = x;
#pragma unroll
    for (int off = 1; off <= 8; off <<= 1)
      mx = fmaxf(mx, __shfl_xor(mx, off, 32));
    float e = __expf(x - mx);
    float sm = e;
#pragma unroll
    for (int off = 1; off <= 8; off <<= 1)
      sm += __shfl_xor(sm, off, 32);
    float att = e / sm;
    if (v + 8 * hi < LWORD) colsum += att;  // exclude pad row l=15
  }
  colsum += __shfl_xor(colsum, 16, 32);  // w[m] now on lane m and m+16
  float wv[16];
#pragma unroll
  for (int i = 0; i < 16; ++i) wv[i] = __shfl(colsum, i, 32);

  int rI = rows[move];
  int cI = cols[move];
  int dI = dirs[move];
  int sc = scores[move];
  int sI = sc < 99 ? sc : 99;

  // ---- V tiles (WMMA) contracted with w in-register; fused epilogue ----
  const _Float16* Wv16 = w16 + 2 * DDIM * DDIM;
  const int colb = lane & 15;
  for (int nt = 0; nt < 16; ++nt) {
    float bval = bv[nt * 16 + colb];
    v8f acc = {bval, bval, bval, bval, bval, bval, bval, bval};
#pragma unroll
    for (int kt = 0; kt < 8; ++kt) {
      v16h a = load_a_tile(embB, lane, kt);
      v16h b = load_b_tile_w(Wv16, lane, nt, kt);
      acc = __builtin_amdgcn_wmma_f32_16x16x32_f16(false, a, false, b, (short)0, acc,
                                                   false, false);
    }
    float partial = 0.0f;
#pragma unroll
    for (int v = 0; v < 8; ++v) {
      float wsel = hi ? wv[v + 8] : wv[v];  // compile-time indices, cndmask select
      partial += wsel * acc[v];
    }
    partial += __shfl_xor(partial, 16, 32);  // combine rows 0-7 and 8-15
    if (lane < 16) {
      int dd = nt * 16 + lane;
      out[move * DDIM + dd] = 2.0f * partial
          + row_emb[rI * DDIM + dd] + col_emb[cI * DDIM + dd]
          + dir_emb[dI * DDIM + dd] + score_emb[sI * DDIM + dd];
    }
  }
}

extern "C" void kernel_launch(void* const* d_in, const int* in_sizes, int n_in,
                              void* d_out, int out_size, void* d_ws, size_t ws_size,
                              hipStream_t stream) {
  (void)in_sizes; (void)n_in; (void)out_size; (void)ws_size;
  const int*   words      = (const int*)d_in[0];
  const int*   rows       = (const int*)d_in[1];
  const int*   cols       = (const int*)d_in[2];
  const int*   dirs       = (const int*)d_in[3];
  const int*   scores     = (const int*)d_in[4];
  const float* letter_emb = (const float*)d_in[5];
  const float* positional = (const float*)d_in[6];
  const float* Wq         = (const float*)d_in[7];
  const float* bq         = (const float*)d_in[8];
  const float* Wk         = (const float*)d_in[9];
  const float* bk         = (const float*)d_in[10];
  const float* Wv         = (const float*)d_in[11];
  const float* bv         = (const float*)d_in[12];
  const float* row_emb    = (const float*)d_in[13];
  const float* col_emb    = (const float*)d_in[14];
  const float* dir_emb    = (const float*)d_in[15];
  const float* score_emb  = (const float*)d_in[16];
  float* out = (float*)d_out;
  _Float16* w16 = (_Float16*)d_ws;  // needs 3*256*256*2 = 384 KB of workspace

  prep_weights_f16<<<(DDIM * DDIM + 255) / 256, 256, 0, stream>>>(Wq, Wk, Wv, w16);

  size_t shmem = (size_t)WPB * 3 * BUFH * sizeof(_Float16);  // 202752 B / block
  move_encoder_kernel<<<NMOVES / WPB, 32 * WPB, shmem, stream>>>(
      words, rows, cols, dirs, scores, letter_emb, positional,
      bq, bk, bv, row_emb, col_emb, dir_emb, score_emb, w16, out);
}
